// SelfAttn_446676599337
// MI455X (gfx1250) — matrix operand order repaired and checked by reading, not compiled, not run on hardware
//
#include <hip/hip_runtime.h>
#include <hip/hip_bf16.h>

typedef __bf16 bf16;
typedef __attribute__((ext_vector_type(8)))  __bf16 v8bf;
typedef __attribute__((ext_vector_type(16))) __bf16 v16bf;
typedef __attribute__((ext_vector_type(8)))  float  v8f;

// ---------------- helpers ----------------
__device__ __forceinline__ bf16 f2bf(float f) {
  union { float f; unsigned u; } a; a.f = f;
  unsigned r = a.u + 0x7FFFu + ((a.u >> 16) & 1u);   // round-to-nearest-even
  unsigned short h = (unsigned short)(r >> 16);
  bf16 out; __builtin_memcpy(&out, &h, 2);
  return out;
}

// A-fragment (16x32 bf16, M x K): lane L holds row m=L%16; VGPR layout:
//   k = (L/16)*16 + j  -> one contiguous 16-elem run per lane.
__device__ __forceinline__ v16bf load_afrag(const bf16* __restrict__ base, int lh) {
  v8bf lo = *reinterpret_cast<const v8bf*>(base + lh * 16);
  v8bf hi = *reinterpret_cast<const v8bf*>(base + lh * 16 + 8);
  return __builtin_shufflevector(lo, hi, 0,1,2,3,4,5,6,7,8,9,10,11,12,13,14,15);
}

// ---------------- fp32 -> bf16 convert ----------------
__global__ void cvt_kernel(const float* __restrict__ in, bf16* __restrict__ out, int n) {
  int i = blockIdx.x * blockDim.x + threadIdx.x;
  if (i < n) out[i] = f2bf(in[i]);
}

// ---------------- LayerNorm (F=1024), bf16 out ----------------
__global__ __launch_bounds__(256) void layernorm_kernel(
    const float* __restrict__ x, const float* __restrict__ wgt,
    const float* __restrict__ bsh, bf16* __restrict__ out) {
  const int row = blockIdx.x;
  const float* xr = x + (size_t)row * 1024;
  float v[4], s = 0.f, sq = 0.f;
#pragma unroll
  for (int i = 0; i < 4; ++i) {
    v[i] = xr[threadIdx.x + i * 256];
    s += v[i]; sq += v[i] * v[i];
  }
#pragma unroll
  for (int m = 16; m >= 1; m >>= 1) { s += __shfl_xor(s, m, 32); sq += __shfl_xor(sq, m, 32); }
  __shared__ float ss[8], ssq[8];
  if ((threadIdx.x & 31) == 0) { ss[threadIdx.x >> 5] = s; ssq[threadIdx.x >> 5] = sq; }
  __syncthreads();
  float ts = 0.f, tsq = 0.f;
#pragma unroll
  for (int i = 0; i < 8; ++i) { ts += ss[i]; tsq += ssq[i]; }
  float mu   = ts  * (1.0f / 1024.0f);
  float var  = tsq * (1.0f / 1024.0f) - mu * mu;
  float rstd = rsqrtf(var + 1e-6f);
#pragma unroll
  for (int i = 0; i < 4; ++i) {
    int c = threadIdx.x + i * 256;
    out[(size_t)row * 1024 + c] = f2bf((v[i] - mu) * rstd * wgt[c] + bsh[c]);
  }
}

// ---------------- WMMA GEMM: C[M,N] = A[M,K] @ W[N,K]^T + bias ----------------
// Block = 8 waves as 2(M) x 4(N); wave tile = 32(M) x 64(N) -> block tile 64x256.
// Register blocking: each B-fragment feeds 2 WMMAs (24 B/WMMA/lane from L2).
// RELU: relu epilogue, bf16 out.  RES: add f32 residual, f32 out.
template <int RELU, int RES>
__global__ __launch_bounds__(256) void gemm_bf16_kernel(
    const bf16* __restrict__ A, const bf16* __restrict__ W,
    const float* __restrict__ bias, const float* __restrict__ residual,
    float* __restrict__ outF, bf16* __restrict__ outB, int K, int N) {
  const int lane = threadIdx.x & 31;
  const int w    = threadIdx.x >> 5;
  const int ln16 = lane & 15;
  const int lh   = lane >> 4;
  const int m0 = blockIdx.x * 64  + (w & 1)  * 32;   // wave M base (2 subtiles of 16)
  const int n0 = blockIdx.y * 256 + (w >> 1) * 64;   // wave N base (4 subtiles of 16)

  v8f acc[2][4] = {};
  const bf16* arow0 = A + (size_t)(m0 + ln16) * K;
  const bf16* arow1 = arow0 + (size_t)16 * K;
  for (int k0 = 0; k0 < K; k0 += 32) {
    v16bf a0 = load_afrag(arow0 + k0, lh);
    v16bf a1 = load_afrag(arow1 + k0, lh);
#pragma unroll
    for (int nt = 0; nt < 4; ++nt) {
      const bf16* wrow = W + (size_t)(n0 + nt * 16 + ln16) * K + k0 + lh * 16;
      v16bf b = *reinterpret_cast<const v16bf*>(wrow);  // B-frag: k = (L/16)*16 + j, contiguous
      acc[0][nt] = __builtin_amdgcn_wmma_f32_16x16x32_bf16(
          false, a0, false, b, (short)0, acc[0][nt], false, false);
      acc[1][nt] = __builtin_amdgcn_wmma_f32_16x16x32_bf16(
          false, a1, false, b, (short)0, acc[1][nt], false, false);
    }
  }
  // Epilogue. C layout: lane holds col n = L%16, rows m = (L/16)*8 + r.
#pragma unroll
  for (int mt = 0; mt < 2; ++mt) {
#pragma unroll
    for (int nt = 0; nt < 4; ++nt) {
      const int col = n0 + nt * 16 + ln16;
      const float bv = bias[col];
#pragma unroll
      for (int r = 0; r < 8; ++r) {
        const int mrow = m0 + mt * 16 + lh * 8 + r;
        float v = acc[mt][nt][r] + bv;
        if (RELU) v = fmaxf(v, 0.0f);
        if (RES) outF[(size_t)mrow * N + col] = v + residual[(size_t)mrow * N + col];
        else     outB[(size_t)mrow * N + col] = f2bf(v);
      }
    }
  }
}

// ---------------- Flash attention (non-causal), WMMA bf16 ----------------
// grid.x = 1024 blocks * 8 waves; wave = one 16-row q-tile; every block's 8
// waves share the same (b,h) so V^T staging in LDS is shared.
#define WPB  8
#define VTS  48   // V^T row stride (bf16 units): rows 96B -> 32B-aligned chunks
#define PSTR 40   // P row stride  (bf16 units): rows 80B -> 16B-aligned chunks

__global__ __launch_bounds__(256) void attn_kernel(
    const bf16* __restrict__ Q, const bf16* __restrict__ Kb,
    const bf16* __restrict__ V, bf16* __restrict__ O) {
  __shared__ bf16 Vt[64 * VTS];            // V^T tile: [d=64][s=32]
  __shared__ bf16 Pl[WPB][16 * PSTR];      // per-wave P tile: [q=16][s=32]
  const int lane = threadIdx.x & 31;
  const int w    = threadIdx.x >> 5;
  const int ln16 = lane & 15;
  const int lh   = lane >> 4;

  const int t  = blockIdx.x * WPB + w;
  const int qt = t & 127;
  const int hh = (t >> 7) & 15;
  const int b  = t >> 11;
  const int q0 = qt * 16;

  const size_t rowQ = ((size_t)b * 2048 + q0 + ln16) * 1024 + hh * 64;
  v16bf aQ0 = load_afrag(Q + rowQ + 0,  lh);   // d = 0..31
  v16bf aQ1 = load_afrag(Q + rowQ + 32, lh);   // d = 32..63

  v8f acc[4] = {};
  float mi[8], li[8];
#pragma unroll
  for (int r = 0; r < 8; ++r) { mi[r] = -1e30f; li[r] = 0.f; }

  const int  sRow  = threadIdx.x >> 3;        // 0..31 : staged s row
  const int  dCh   = (threadIdx.x & 7) * 8;   // 0..56 : staged d chunk
  const bf16* vbase = V + ((size_t)b * 2048) * 1024 + hh * 64;

  for (int s0 = 0; s0 < 2048; s0 += 32) {
    // cooperative V^T staging (coalesced read, transposed LDS write)
    v8bf vv = *reinterpret_cast<const v8bf*>(vbase + (size_t)(s0 + sRow) * 1024 + dCh);
#pragma unroll
    for (int i = 0; i < 8; ++i) Vt[(dCh + i) * VTS + sRow] = vv[i];
    __syncthreads();

    // scores S = Q K^T for 32 keys (two 16-col frags, d contracted in 2 steps)
    v8f sc0 = {}, sc1 = {};
    {
      const bf16* kr0 = Kb + ((size_t)b * 2048 + s0 + ln16) * 1024 + hh * 64 + lh * 16;
      v16bf bk = *reinterpret_cast<const v16bf*>(kr0);
      sc0 = __builtin_amdgcn_wmma_f32_16x16x32_bf16(false, aQ0, false, bk, (short)0, sc0, false, false);
      bk = *reinterpret_cast<const v16bf*>(kr0 + 32);
      sc0 = __builtin_amdgcn_wmma_f32_16x16x32_bf16(false, aQ1, false, bk, (short)0, sc0, false, false);
      const bf16* kr1 = Kb + ((size_t)b * 2048 + s0 + 16 + ln16) * 1024 + hh * 64 + lh * 16;
      v16bf bk1 = *reinterpret_cast<const v16bf*>(kr1);
      sc1 = __builtin_amdgcn_wmma_f32_16x16x32_bf16(false, aQ0, false, bk1, (short)0, sc1, false, false);
      bk1 = *reinterpret_cast<const v16bf*>(kr1 + 32);
      sc1 = __builtin_amdgcn_wmma_f32_16x16x32_bf16(false, aQ1, false, bk1, (short)0, sc1, false, false);
    }

    // online softmax: row = lh*8 + r spans the 16 lanes of one half-wave
    float alpha[8];
#pragma unroll
    for (int r = 0; r < 8; ++r) {
      float v0 = sc0[r] * 0.125f, v1 = sc1[r] * 0.125f;   // 1/sqrt(64)
      float mx = fmaxf(v0, v1);
#pragma unroll
      for (int m = 8; m >= 1; m >>= 1) mx = fmaxf(mx, __shfl_xor(mx, m, 32));
      float mn = fmaxf(mi[r], mx);
      alpha[r] = __expf(mi[r] - mn);
      float p0 = __expf(v0 - mn), p1 = __expf(v1 - mn);
      float rs = p0 + p1;
#pragma unroll
      for (int m = 8; m >= 1; m >>= 1) rs += __shfl_xor(rs, m, 32);
      li[r] = li[r] * alpha[r] + rs;
      mi[r] = mn;
      const int prow = lh * 8 + r;
      Pl[w][prow * PSTR + ln16]      = f2bf(p0);
      Pl[w][prow * PSTR + 16 + ln16] = f2bf(p1);
#pragma unroll
      for (int nt = 0; nt < 4; ++nt) acc[nt][r] *= alpha[r];
    }
    __syncthreads();

    // O += P @ V  (contract s=32; A-frag from P in LDS, B-frags from V^T)
    v16bf aP;
    {
      const bf16* pr = &Pl[w][ln16 * PSTR + lh * 16];
      v8bf lo = *reinterpret_cast<const v8bf*>(pr);
      v8bf hi = *reinterpret_cast<const v8bf*>(pr + 8);
      aP = __builtin_shufflevector(lo, hi, 0,1,2,3,4,5,6,7,8,9,10,11,12,13,14,15);
    }
#pragma unroll
    for (int nt = 0; nt < 4; ++nt) {
      const bf16* vr = &Vt[(nt * 16 + ln16) * VTS + lh * 16];
      v16bf bv = *reinterpret_cast<const v16bf*>(vr);
      acc[nt] = __builtin_amdgcn_wmma_f32_16x16x32_bf16(
          false, aP, false, bv, (short)0, acc[nt], false, false);
    }
    __syncthreads();   // all Vt readers done before next stage
  }

  // normalize and store: O[b, q0+m, hh*64 + d]
#pragma unroll
  for (int nt = 0; nt < 4; ++nt) {
#pragma unroll
    for (int r = 0; r < 8; ++r) {
      const int mrow = q0 + lh * 8 + r;
      float ov = acc[nt][r] / li[r];
      O[((size_t)b * 2048 + mrow) * 1024 + hh * 64 + nt * 16 + ln16] = f2bf(ov);
    }
  }
}

// ---------------- host launch ----------------
extern "C" void kernel_launch(void* const* d_in, const int* in_sizes, int n_in,
                              void* d_out, int out_size, void* d_ws, size_t ws_size,
                              hipStream_t stream) {
  (void)in_sizes; (void)n_in; (void)out_size; (void)ws_size;
  const float* x     = (const float*)d_in[0];
  const float* ln1w  = (const float*)d_in[1];
  const float* ln1b  = (const float*)d_in[2];
  const float* wq_w  = (const float*)d_in[3];
  const float* wq_b  = (const float*)d_in[4];
  const float* wk_w  = (const float*)d_in[5];
  const float* wk_b  = (const float*)d_in[6];
  const float* wv_w  = (const float*)d_in[7];
  const float* wv_b  = (const float*)d_in[8];
  const float* fc_w  = (const float*)d_in[9];
  const float* fc_b  = (const float*)d_in[10];
  const float* ln2w  = (const float*)d_in[11];
  const float* ln2b  = (const float*)d_in[12];
  const float* fc1_w = (const float*)d_in[13];
  const float* fc1_b = (const float*)d_in[14];
  const float* fc2_w = (const float*)d_in[15];
  const float* fc2_b = (const float*)d_in[16];

  const size_t MB = 1024ull * 1024ull;
  char* ws = (char*)d_ws;
  bf16* wq_bf  = (bf16*)(ws + 0 * MB);     // 1024x1024 bf16 = 2MB
  bf16* wk_bf  = (bf16*)(ws + 2 * MB);
  bf16* wv_bf  = (bf16*)(ws + 4 * MB);
  bf16* fcw_bf = (bf16*)(ws + 6 * MB);
  bf16* fc1_bf = (bf16*)(ws + 8 * MB);     // 4096x1024 bf16 = 8MB
  bf16* fc2_bf = (bf16*)(ws + 16 * MB);    // 1024x4096 bf16 = 8MB
  bf16* h_bf   = (bf16*)(ws + 24 * MB);    // 8192x1024 bf16 = 16MB
  bf16* q_bf   = (bf16*)(ws + 40 * MB);
  bf16* k_bf   = (bf16*)(ws + 56 * MB);
  bf16* v_bf   = (bf16*)(ws + 72 * MB);
  bf16* att_bf = (bf16*)(ws + 88 * MB);
  float* xmid  = (float*)(ws + 104 * MB);  // 8192x1024 f32 = 32MB
  bf16* h2_bf  = (bf16*)(ws + 24 * MB);    // reuse h (dead after QKV)
  bf16* ff_bf  = (bf16*)(ws + 40 * MB);    // 8192x4096 bf16 = 64MB, reuse q/k/v/att

  auto cvt = [&](const float* src, bf16* dst, int n) {
    cvt_kernel<<<(n + 255) / 256, 256, 0, stream>>>(src, dst, n);
  };
  cvt(wq_w,  wq_bf,  1024 * 1024);
  cvt(wk_w,  wk_bf,  1024 * 1024);
  cvt(wv_w,  wv_bf,  1024 * 1024);
  cvt(fc_w,  fcw_bf, 1024 * 1024);
  cvt(fc1_w, fc1_bf, 4096 * 1024);
  cvt(fc2_w, fc2_bf, 1024 * 4096);

  // LN1
  layernorm_kernel<<<8192, 256, 0, stream>>>(x, ln1w, ln1b, h_bf);

  // QKV projections (block tile 64x256)
  dim3 g1(128, 4);   // M/64 x N/256, N=1024
  gemm_bf16_kernel<0, 0><<<g1, 256, 0, stream>>>(h_bf, wq_bf, wq_b, nullptr, nullptr, q_bf, 1024, 1024);
  gemm_bf16_kernel<0, 0><<<g1, 256, 0, stream>>>(h_bf, wk_bf, wk_b, nullptr, nullptr, k_bf, 1024, 1024);
  gemm_bf16_kernel<0, 0><<<g1, 256, 0, stream>>>(h_bf, wv_bf, wv_b, nullptr, nullptr, v_bf, 1024, 1024);

  // attention: 4 * 16 * 128 q-tiles / 8 waves per block
  attn_kernel<<<1024, 256, 0, stream>>>(q_bf, k_bf, v_bf, att_bf);

  // out-proj + residual -> xmid (f32)
  gemm_bf16_kernel<0, 1><<<g1, 256, 0, stream>>>(att_bf, fcw_bf, fc_b, x, xmid, nullptr, 1024, 1024);

  // LN2
  layernorm_kernel<<<8192, 256, 0, stream>>>(xmid, ln2w, ln2b, h2_bf);

  // MLP up (ReLU) -> ff (bf16), N=4096
  dim3 g2(128, 16);
  gemm_bf16_kernel<1, 0><<<g2, 256, 0, stream>>>(h2_bf, fc1_bf, fc1_b, nullptr, nullptr, ff_bf, 1024, 4096);

  // MLP down + residual -> d_out (f32), K=4096
  gemm_bf16_kernel<0, 1><<<g1, 256, 0, stream>>>(ff_bf, fc2_bf, fc2_b, xmid, (float*)d_out, nullptr, 4096, 1024);
}
